// MultiHeadAttentionLayer_66340064854782
// MI455X (gfx1250) — compile-verified, matrix operand order checked by reading, not run on hardware
//
#include <hip/hip_runtime.h>

#define S_LEN 1024
#define EMB   1024
#define NH    16
#define HD    64

typedef __attribute__((ext_vector_type(16))) __bf16 v16bf;
typedef __attribute__((ext_vector_type(8)))  __bf16 v8bf;
typedef __attribute__((ext_vector_type(4)))  __bf16 v4bf;
typedef __attribute__((ext_vector_type(8)))  float  v8f;
typedef __attribute__((ext_vector_type(4)))  float  v4f;
typedef __attribute__((ext_vector_type(4)))  int    v4i;

#if __has_builtin(__builtin_amdgcn_global_load_async_to_lds_b128)
#define HAVE_ASYNC_LDS 1
#else
#define HAVE_ASYNC_LDS 0
#endif

#if HAVE_ASYNC_LDS
typedef __attribute__((address_space(1))) v4i* gp128_t;  // global b128 chunk
typedef __attribute__((address_space(3))) v4i* lp128_t;  // LDS b128 chunk
static __device__ __forceinline__ void async_copy_b128(const __bf16* g, __bf16* l) {
  __bf16* gnc = const_cast<__bf16*>(g);
  __builtin_amdgcn_global_load_async_to_lds_b128((gp128_t)gnc, (lp128_t)l, 0, 0);
}
template <int N>
static __device__ __forceinline__ void wait_async_le() {
#if __has_builtin(__builtin_amdgcn_s_wait_asynccnt)
  __builtin_amdgcn_s_wait_asynccnt(N);
#else
  asm volatile("s_wait_asynccnt %0" ::"i"(N) : "memory");
#endif
}
#endif

static __device__ __forceinline__ v8f zero8() {
  v8f z;
#pragma unroll
  for (int j = 0; j < 8; ++j) z[j] = 0.0f;
  return z;
}

static __device__ __forceinline__ v8f wmma_bf16(v16bf a, v16bf b, v8f c) {
  // D = A(16x32) * B(32x16) + C, fp32 accumulate
  return __builtin_amdgcn_wmma_f32_16x16x32_bf16(false, a, false, b,
                                                 (short)0, c, false, false);
}

// NT fragment load: operand row-major [rows x ld] bf16, K contiguous per lane.
// Lane L: row = row0 + (L&15); 8-elem chunks at k0+(L>>4)*8 and k0+16+(L>>4)*8.
static __device__ __forceinline__ v16bf load_frag(const __bf16* __restrict__ base,
                                                  int row0, int k0, int ld) {
  const int lane = threadIdx.x & 31;
  const int r  = row0 + (lane & 15);
  const int kc = k0 + ((lane >> 4) << 3);
  union { v16bf v; v8bf h[2]; } u;
  u.h[0] = *reinterpret_cast<const v8bf*>(base + (size_t)r * ld + kc);
  u.h[1] = *reinterpret_cast<const v8bf*>(base + (size_t)r * ld + kc + 16);
  return u.v;
}

// --- fp32 -> bf16 convert (vectorized) -------------------------------------
__global__ void cvt_f32_bf16(const float* __restrict__ s, __bf16* __restrict__ d, int n4) {
  int i = blockIdx.x * blockDim.x + threadIdx.x;
  if (i < n4) {
    v4f x = *reinterpret_cast<const v4f*>(s + 4 * (size_t)i);
    v4bf y;
#pragma unroll
    for (int j = 0; j < 4; ++j) y[j] = (__bf16)x[j];
    *reinterpret_cast<v4bf*>(d + 4 * (size_t)i) = y;
  }
}

// --- NT GEMM: C[M,N] = A[M,K] * B[N,K]^T + bias ----------------------------
// LDS-staged, double-buffered, async-copy pipelined.
// MODE 0: bf16 permuted to Qh  [B,H,S,D]
// MODE 1: bf16 transposed to Vt [B,H,D,S]
// MODE 2: fp32 row-major [M,N]
template <int MODE>
__global__ __launch_bounds__(256, 2)
void gemm_nt(const __bf16* __restrict__ A, const __bf16* __restrict__ B,
             const float* __restrict__ bias, void* __restrict__ out,
             int M, int N, int K) {
  __shared__ __align__(16) __bf16 As[2][128][72];  // pad 72 to spread LDS banks
  __shared__ __align__(16) __bf16 Bs[2][128][72];

  const int tid  = threadIdx.x;
  const int lane = tid & 31;
  const int wave = tid >> 5;
  const int wm = wave & 3, wn = wave >> 2;         // wave tile: 32 rows x 64 cols
  const int bm = blockIdx.x * 128, bn = blockIdx.y * 128;

  // staging: each thread copies 32 contiguous bf16 (4 x b128) per matrix/stage
  const int srow = tid & 127;
  const int scol = (tid >> 7) << 5;                // 0 or 32
  const __bf16* gA = A + (size_t)(bm + srow) * K + scol;
  const __bf16* gB = B + (size_t)(bn + srow) * K + scol;

  v8f acc[2][4];
#pragma unroll
  for (int i = 0; i < 2; ++i)
#pragma unroll
    for (int j = 0; j < 4; ++j) acc[i][j] = zero8();

  const int nstage = K >> 6;  // K/64

  auto compute = [&](int bufi) {
#pragma unroll
    for (int kk = 0; kk < 64; kk += 32) {
      v16bf af[2], bfr[4];
#pragma unroll
      for (int mt = 0; mt < 2; ++mt)
        af[mt] = load_frag(&As[bufi][0][0], wm * 32 + mt * 16, kk, 72);
#pragma unroll
      for (int nt = 0; nt < 4; ++nt)
        bfr[nt] = load_frag(&Bs[bufi][0][0], wn * 64 + nt * 16, kk, 72);
#pragma unroll
      for (int mt = 0; mt < 2; ++mt)
#pragma unroll
        for (int nt = 0; nt < 4; ++nt)
          acc[mt][nt] = wmma_bf16(af[mt], bfr[nt], acc[mt][nt]);
    }
  };

#if HAVE_ASYNC_LDS
  auto issue = [&](int ks) {
    const int bufi = ks & 1;
    const __bf16* a = gA + (ks << 6);
    const __bf16* b = gB + (ks << 6);
    __bf16* la = &As[bufi][srow][scol];
    __bf16* lb = &Bs[bufi][srow][scol];
#pragma unroll
    for (int c = 0; c < 4; ++c) {
      async_copy_b128(a + c * 8, la + c * 8);
      async_copy_b128(b + c * 8, lb + c * 8);
    }
  };
  issue(0);
  for (int ks = 0; ks < nstage; ++ks) {
    if (ks + 1 < nstage) {
      issue(ks + 1);          // DMA next stage while we compute this one
      wait_async_le<8>();     // stage ks complete (<= 8 newer ops in flight)
    } else {
      wait_async_le<0>();
    }
    __syncthreads();
    compute(ks & 1);
    __syncthreads();
  }
#else
  v8bf ra[4], rb[4];
  auto gload = [&](int ks) {
    const __bf16* a = gA + (ks << 6);
    const __bf16* b = gB + (ks << 6);
#pragma unroll
    for (int c = 0; c < 4; ++c) {
      ra[c] = *reinterpret_cast<const v8bf*>(a + c * 8);
      rb[c] = *reinterpret_cast<const v8bf*>(b + c * 8);
    }
  };
  gload(0);
  for (int ks = 0; ks < nstage; ++ks) {
    const int bufi = ks & 1;
#pragma unroll
    for (int c = 0; c < 4; ++c) {
      *reinterpret_cast<v8bf*>(&As[bufi][srow][scol + c * 8]) = ra[c];
      *reinterpret_cast<v8bf*>(&Bs[bufi][srow][scol + c * 8]) = rb[c];
    }
    __syncthreads();
    if (ks + 1 < nstage) gload(ks + 1);  // overlap global loads with WMMAs
    compute(bufi);
    __syncthreads();
  }
#endif

  const int half = lane >> 4, nl = lane & 15;
#pragma unroll
  for (int mt = 0; mt < 2; ++mt)
#pragma unroll
    for (int nt = 0; nt < 4; ++nt) {
      const int c = bn + wn * 64 + nt * 16 + nl;
      const float bv = bias[c];
#pragma unroll
      for (int j = 0; j < 8; ++j) {
        const int r = bm + wm * 32 + mt * 16 + (half << 3) + j;  // C row = j+8*half
        const float v = acc[mt][nt][j] + bv;
        if (MODE == 2) {
          ((float*)out)[(size_t)r * N + c] = v;
        } else {
          const int bb = r >> 10, ss = r & 1023;  // r = b*S + s
          const int hh = c >> 6,  dd = c & 63;    // c = h*D + d
          if (MODE == 0)
            ((__bf16*)out)[((size_t)(bb * NH + hh) * S_LEN + ss) * HD + dd] = (__bf16)v;
          else
            ((__bf16*)out)[((size_t)(bb * NH + hh) * HD + dd) * S_LEN + ss] = (__bf16)v;
        }
      }
    }
}

// --- cross-lane reductions within 16-lane halves (wave32) ------------------
static __device__ __forceinline__ float redmax16(float v) {
  v = fmaxf(v, __shfl_xor(v, 1, 32));
  v = fmaxf(v, __shfl_xor(v, 2, 32));
  v = fmaxf(v, __shfl_xor(v, 4, 32));
  v = fmaxf(v, __shfl_xor(v, 8, 32));
  return v;
}
static __device__ __forceinline__ float redsum16(float v) {
  v += __shfl_xor(v, 1, 32);
  v += __shfl_xor(v, 2, 32);
  v += __shfl_xor(v, 4, 32);
  v += __shfl_xor(v, 8, 32);
  return v;
}

// --- fused attention: logits WMMA + rel-bias + online softmax + PV WMMA ----
// Pair-structured: two 16-wide K-tiles per iteration, one joint max/rescale,
// one K=32 PV WMMA step. K-frags and V-frags prefetched a full pair ahead.
// Qh: [B*H, S, D] bf16 ; Vt: [B*H, D, S] bf16 ; rpk/rpv: [5,64] fp32
// O1: [B*S, E] bf16
__global__ __launch_bounds__(128)
void attn_fused(const __bf16* __restrict__ Qh, const __bf16* __restrict__ Vt,
                const float* __restrict__ rpk, const float* __restrict__ rpv,
                __bf16* __restrict__ O1) {
  const int lane = threadIdx.x & 31;
  const int wv   = threadIdx.x >> 5;            // 4 independent waves / block
  const int bh   = blockIdx.x;                  // b*16 + h
  const int b    = bh >> 4, h = bh & 15;
  const int q0   = (blockIdx.y * 4 + wv) << 4;  // 16-row query tile
  const int half = lane >> 4, nl = lane & 15, myrow = half << 3;

  const __bf16* Qb = Qh + (size_t)bh * S_LEN * HD;
  const __bf16* Vb = Vt + (size_t)bh * HD * S_LEN;

  __shared__ float r_s[4][80];                      // 5 rel-dots x 16 rows / wave
  __shared__ __align__(16) __bf16 pbuf[4][16][32];  // p-pair staging / wave

  // r_s[t][m] = (qh[q0+m] . rpk[t]) / 8
  if (lane < 16) {
#pragma unroll
    for (int t = 0; t < 5; ++t) {
      float a = 0.0f;
      for (int d = 0; d < HD; ++d)
        a += (float)Qb[(size_t)(q0 + lane) * HD + d] * rpk[t * HD + d];
      r_s[wv][t * 16 + lane] = a * 0.125f;
    }
  }
  __syncthreads();

  float r4v[8], r0v[8];
#pragma unroll
  for (int j = 0; j < 8; ++j) {
    r4v[j] = r_s[wv][64 + myrow + j];   // t=4 (prefix band)
    r0v[j] = r_s[wv][myrow + j];        // t=0 (suffix band)
  }

  const v16bf qa0 = load_frag(Qb, q0, 0, HD);
  const v16bf qa1 = load_frag(Qb, q0, 32, HD);

  v8f oacc[4];
#pragma unroll
  for (int nt = 0; nt < 4; ++nt) oacc[nt] = zero8();
  float mrun[8], lrun[8], s1[8], s2[8], s3[8], s4[8];
#pragma unroll
  for (int j = 0; j < 8; ++j) {
    mrun[j] = -1e30f; lrun[j] = 0.f;
    s1[j] = s2[j] = s3[j] = s4[j] = 0.f;
  }

  // rel-position bias add (scaled); t = clip(q-k,-2,2)+2
  auto relbias = [&](v8f& lg, int tk0) {
    const bool pre = (tk0 + 32 <= q0);
    const bool suf = (tk0 >= q0 + 32);
    if (pre) {
#pragma unroll
      for (int j = 0; j < 8; ++j) lg[j] = lg[j] * 0.125f + r4v[j];
    } else if (suf) {
#pragma unroll
      for (int j = 0; j < 8; ++j) lg[j] = lg[j] * 0.125f + r0v[j];
    } else {
#pragma unroll
      for (int j = 0; j < 8; ++j) {
        int dd = (q0 + myrow + j) - (tk0 + nl);
        int t = (dd < -2 ? -2 : (dd > 2 ? 2 : dd)) + 2;
        lg[j] = lg[j] * 0.125f + r_s[wv][t * 16 + myrow + j];
      }
    }
  };
  // banded masses for the 3 diagonal-adjacent tiles
  auto band = [&](const float* p, int tk0) {
#pragma unroll
    for (int j = 0; j < 8; ++j) {
      const int dd = (q0 + myrow + j) - (tk0 + nl);
      const float pv = p[j];
      s4[j] += redsum16(dd >= 2 ? pv : 0.f);
      s3[j] += redsum16(dd == 1 ? pv : 0.f);
      s2[j] += redsum16(dd == 0 ? pv : 0.f);
      s1[j] += redsum16(dd == -1 ? pv : 0.f);
    }
  };

  // pipeline preload: K-frags + V-frags for pair 0
  v16bf kb[4], vf[4];
  kb[0] = load_frag(Qb, 0, 0, HD);  kb[1] = load_frag(Qb, 0, 32, HD);
  kb[2] = load_frag(Qb, 16, 0, HD); kb[3] = load_frag(Qb, 16, 32, HD);
#pragma unroll
  for (int nt = 0; nt < 4; ++nt) vf[nt] = load_frag(Vb, nt * 16, 0, S_LEN);

  for (int kp = 0; kp < S_LEN / 32; ++kp) {
    const int k0 = kp << 5;                    // tiles at k0 and k0+16

    v8f lgA = zero8(), lgB = zero8();
    lgA = wmma_bf16(qa0, kb[0], lgA);
    lgA = wmma_bf16(qa1, kb[1], lgA);
    lgB = wmma_bf16(qa0, kb[2], lgB);
    lgB = wmma_bf16(qa1, kb[3], lgB);

    if (kp + 1 < S_LEN / 32) {                 // prefetch next pair's K-frags
      kb[0] = load_frag(Qb, k0 + 32, 0, HD);
      kb[1] = load_frag(Qb, k0 + 32, 32, HD);
      kb[2] = load_frag(Qb, k0 + 48, 0, HD);
      kb[3] = load_frag(Qb, k0 + 48, 32, HD);
    }

    relbias(lgA, k0);
    relbias(lgB, k0 + 16);

    // joint online softmax over the 32-column pair (one rescale per pair)
    float pA[8], pB[8], rsA[8], rsB[8];
#pragma unroll
    for (int j = 0; j < 8; ++j) {
      const float tm = fmaxf(redmax16(lgA[j]), redmax16(lgB[j]));
      const float mn = fmaxf(mrun[j], tm);
      const float al = __expf(mrun[j] - mn);
      mrun[j] = mn;
      lrun[j] *= al; s1[j] *= al; s2[j] *= al; s3[j] *= al; s4[j] *= al;
      oacc[0][j] *= al; oacc[1][j] *= al; oacc[2][j] *= al; oacc[3][j] *= al;
      pA[j] = __expf(lgA[j] - mn);
      pB[j] = __expf(lgB[j] - mn);
      rsA[j] = redsum16(pA[j]);
      rsB[j] = redsum16(pB[j]);
      lrun[j] += rsA[j] + rsB[j];
    }

    const bool preA = (k0 + 32 <= q0), sufA = (k0 >= q0 + 32);
    const bool preB = (k0 + 48 <= q0), sufB = (k0 + 16 >= q0 + 32);
    if (preA) {
#pragma unroll
      for (int j = 0; j < 8; ++j) s4[j] += rsA[j];
    } else if (!sufA) {
      band(pA, k0);
    }
    if (preB) {
#pragma unroll
      for (int j = 0; j < 8; ++j) s4[j] += rsB[j];
    } else if (!sufB) {
      band(pB, k0 + 16);
    }

    // stage p-pair (bf16) in LDS, re-fragment as WMMA A operand
#pragma unroll
    for (int j = 0; j < 8; ++j) {
      pbuf[wv][myrow + j][nl]      = (__bf16)pA[j];
      pbuf[wv][myrow + j][16 + nl] = (__bf16)pB[j];
    }
    union { v16bf v; v8bf h2[2]; } ua;
    const int rr = nl, cc = half << 3;
    ua.h2[0] = *reinterpret_cast<const v8bf*>(&pbuf[wv][rr][cc]);
    ua.h2[1] = *reinterpret_cast<const v8bf*>(&pbuf[wv][rr][cc + 16]);
#pragma unroll
    for (int nt = 0; nt < 4; ++nt)
      oacc[nt] = wmma_bf16(ua.v, vf[nt], oacc[nt]);

    if (kp + 1 < S_LEN / 32) {                 // prefetch next pair's V-frags
#pragma unroll
      for (int nt = 0; nt < 4; ++nt)
        vf[nt] = load_frag(Vb, nt * 16, k0 + 32, S_LEN);
    }
  }

  // epilogue: normalize, add banded rpv correction, store bf16 [B*S, E]
  float inv[8], c0[8], c1[8], c2[8], c3[8], c4[8];
#pragma unroll
  for (int j = 0; j < 8; ++j) {
    inv[j] = 1.0f / lrun[j];
    c4[j] = s4[j] * inv[j]; c3[j] = s3[j] * inv[j];
    c2[j] = s2[j] * inv[j]; c1[j] = s1[j] * inv[j];
    c0[j] = 1.0f - (c1[j] + c2[j] + c3[j] + c4[j]);
  }
#pragma unroll
  for (int nt = 0; nt < 4; ++nt) {
    const int d = nt * 16 + nl;
    const float rv0 = rpv[d],       rv1 = rpv[64 + d], rv2 = rpv[128 + d];
    const float rv3 = rpv[192 + d], rv4 = rpv[256 + d];
#pragma unroll
    for (int j = 0; j < 8; ++j) {
      const int q = q0 + myrow + j;
      const float val = oacc[nt][j] * inv[j]
                      + c0[j] * rv0 + c1[j] * rv1 + c2[j] * rv2
                      + c3[j] * rv3 + c4[j] * rv4;
      O1[((size_t)(b * S_LEN + q)) * EMB + h * HD + d] = (__bf16)val;
    }
  }
}

// ---------------------------------------------------------------------------
extern "C" void kernel_launch(void* const* d_in, const int* in_sizes, int n_in,
                              void* d_out, int out_size, void* d_ws, size_t ws_size,
                              hipStream_t stream) {
  (void)in_sizes; (void)n_in; (void)out_size; (void)ws_size;
  const float* query = (const float*)d_in[0];
  // d_in[1] = key : unused (reference computes the k-projection but never uses it)
  const float* value = (const float*)d_in[2];
  const float* Wq = (const float*)d_in[3];
  const float* bq = (const float*)d_in[4];
  // d_in[5], d_in[6] = Wk, bk : unused
  const float* Wv = (const float*)d_in[7];
  const float* bv = (const float*)d_in[8];
  const float* Wo = (const float*)d_in[9];
  const float* bo = (const float*)d_in[10];
  const float* rpk = (const float*)d_in[11];
  const float* rpv = (const float*)d_in[12];

  char* ws = (char*)d_ws;
  const size_t MB = 1024 * 1024;
  __bf16* Xq  = (__bf16*)(ws + 0 * MB);    // 8 MB  query bf16 [4096,1024]
  __bf16* Xv  = (__bf16*)(ws + 8 * MB);    // 8 MB  value bf16
  __bf16* Wqb = (__bf16*)(ws + 16 * MB);   // 2 MB
  __bf16* Wvb = (__bf16*)(ws + 18 * MB);   // 2 MB
  __bf16* Wob = (__bf16*)(ws + 20 * MB);   // 2 MB
  __bf16* Qh  = (__bf16*)(ws + 22 * MB);   // 8 MB  [B,H,S,D]
  __bf16* Vt  = (__bf16*)(ws + 30 * MB);   // 8 MB  [B,H,D,S]
  __bf16* O1  = (__bf16*)(ws + 38 * MB);   // 8 MB  [B*S,E]

  const int NTOK = 4096, E = 1024;
  cvt_f32_bf16<<<(NTOK * E / 4 + 255) / 256, 256, 0, stream>>>(query, Xq, NTOK * E / 4);
  cvt_f32_bf16<<<(NTOK * E / 4 + 255) / 256, 256, 0, stream>>>(value, Xv, NTOK * E / 4);
  cvt_f32_bf16<<<(E * E / 4 + 255) / 256, 256, 0, stream>>>(Wq, Wqb, E * E / 4);
  cvt_f32_bf16<<<(E * E / 4 + 255) / 256, 256, 0, stream>>>(Wv, Wvb, E * E / 4);
  cvt_f32_bf16<<<(E * E / 4 + 255) / 256, 256, 0, stream>>>(Wo, Wob, E * E / 4);

  dim3 gg(NTOK / 128, E / 128), bb(256);
  gemm_nt<0><<<gg, bb, 0, stream>>>(Xq, Wqb, bq, Qh, NTOK, E, E);
  gemm_nt<1><<<gg, bb, 0, stream>>>(Xv, Wvb, bv, Vt, NTOK, E, E);

  attn_fused<<<dim3(64, 16), dim3(128), 0, stream>>>(Qh, Vt, rpk, rpv, O1);

  gemm_nt<2><<<gg, bb, 0, stream>>>(O1, Wob, bo, d_out, NTOK, E, E);
}
